// Attention_40819369181401
// MI455X (gfx1250) — matrix operand order repaired and checked by reading, not compiled, not run on hardware
//
#include <hip/hip_runtime.h>
#include <hip/hip_bf16.h>

// Problem constants (match reference)
#define HD     16
#define DMODEL 1024
#define LSEQ   2048
#define BATCH  2

typedef __attribute__((ext_vector_type(16))) __bf16 v16bf;
typedef __attribute__((ext_vector_type(8)))  float  v8f;

union AFrag { v16bf v; unsigned u[8]; };
union CAcc  { v8f  v; float    f[8]; };

__device__ __forceinline__ unsigned short f2bf(float f) {
  union { float f; unsigned u; } x; x.f = f;
  unsigned r = x.u + 0x7FFFu + ((x.u >> 16) & 1u);   // round-to-nearest-even
  return (unsigned short)(r >> 16);
}
__device__ __forceinline__ unsigned pack2(float lo, float hi) {
  return (unsigned)f2bf(lo) | ((unsigned)f2bf(hi) << 16);
}

// ---------------------------------------------------------------------------
// Generic 64x64-tile GEMM: C = A[M=2048,K=1024] x B[K=1024,N=1024] per batch.
// A,B f32 in global; staged to LDS as packed bf16; v_wmma_f32_16x16x32_bf16.
// 128 threads = 4 waves; wave w computes a 16x64 strip (4 C tiles).
// MODE 0: out = bf16 [B,H,M,64]        (value scaled by `scale`)    -> q,k
// MODE 2: out = bf16 [B,H,64,M] (T)                                 -> v
// MODE 1: out = f32  [B,M,1024] sigmoid(acc + cbias[col])           -> gate
// MODE 3: out = f32  [B,M,1024] acc + cbias[col]; A *= G elementwise-> final
// ---------------------------------------------------------------------------
template <int MODE>
__global__ __launch_bounds__(128)
void gemm_proj(const float* __restrict__ A, const float* __restrict__ Bw,
               const float* __restrict__ G, const float* __restrict__ cbias,
               void* __restrict__ out, float scale)
{
  __shared__ unsigned As[64 * 16];   // [row][kpair] bf16x2
  __shared__ unsigned Bs[64 * 16];   // [col][kpair] bf16x2 (B transposed)

  const int b    = blockIdx.z;
  const int row0 = blockIdx.y * 64;
  const int col0 = blockIdx.x * 64;
  const int t    = threadIdx.x;
  const int wave = t >> 5, lane = t & 31;
  const int lh   = lane & 15;
  const int hi   = lane >> 4;
  const int bpA  = hi * 8;          // A-frag pair base (K split 0..15 / 16..31)
  const int bpB  = hi * 8;          // B-frag pair base (K split 0..15 / 16..31)

  CAcc acc[4];
  #pragma unroll
  for (int c = 0; c < 4; ++c)
    #pragma unroll
    for (int v = 0; v < 8; ++v) acc[c].f[v] = 0.0f;

  const int rA = t >> 1;            // 0..63 : A row / B column staged by thread
  const int kh = t & 1;             // which 16-wide K half
  const size_t abase = ((size_t)(b * LSEQ + row0 + rA)) * DMODEL + kh * 16;
  const float* Ap = A + abase;
  const float* Gp = (MODE == 3) ? (G + abase) : nullptr;
  const float* Bp0 = Bw + (size_t)(kh * 16) * DMODEL + col0 + rA;

  for (int k0 = 0; k0 < DMODEL; k0 += 32) {
    // ---- stage A tile (64x32 f32 -> bf16 pairs) ----
    #pragma unroll
    for (int i = 0; i < 8; ++i) {
      float f0 = Ap[k0 + 2 * i], f1 = Ap[k0 + 2 * i + 1];
      if (MODE == 3) { f0 *= Gp[k0 + 2 * i]; f1 *= Gp[k0 + 2 * i + 1]; }
      As[rA * 16 + kh * 8 + i] = pack2(f0, f1);
    }
    // ---- stage B tile transposed (32x64 -> [col][k]) ----
    const float* Bp = Bp0 + (size_t)k0 * DMODEL;
    #pragma unroll
    for (int i = 0; i < 8; ++i)
      Bs[rA * 16 + kh * 8 + i] =
          pack2(Bp[(2 * i) * DMODEL], Bp[(2 * i + 1) * DMODEL]);
    __syncthreads();

    AFrag a;
    #pragma unroll
    for (int j = 0; j < 4; ++j) {
      a.u[j]     = As[(wave * 16 + lh) * 16 + bpA + j];
      a.u[j + 4] = As[(wave * 16 + lh) * 16 + bpA + 4 + j];
    }
    #pragma unroll
    for (int c = 0; c < 4; ++c) {
      AFrag bf;
      #pragma unroll
      for (int j = 0; j < 8; ++j)
        bf.u[j] = Bs[(c * 16 + lh) * 16 + bpB + j];
      acc[c].v = __builtin_amdgcn_wmma_f32_16x16x32_bf16(
          false, a.v, false, bf.v, (short)0, acc[c].v, false, false);
    }
    __syncthreads();
  }

  // ---- epilogue ----
  #pragma unroll
  for (int c = 0; c < 4; ++c) {
    #pragma unroll
    for (int v = 0; v < 8; ++v) {
      const int row = row0 + wave * 16 + v + hi * 8;
      const int col = col0 + c * 16 + lh;
      const float val = acc[c].f[v] * scale;
      if (MODE == 0) {
        ((unsigned short*)out)[(((size_t)(b * HD + (col >> 6)) * LSEQ + row) << 6)
                               + (col & 63)] = f2bf(val);
      } else if (MODE == 2) {
        ((unsigned short*)out)[(((size_t)(b * HD + (col >> 6)) * 64 + (col & 63))
                                * LSEQ) + row] = f2bf(val);
      } else if (MODE == 1) {
        const float g = val + cbias[col];
        ((float*)out)[((size_t)(b * LSEQ + row)) * DMODEL + col] =
            1.0f / (1.0f + __expf(-g));
      } else {
        ((float*)out)[((size_t)(b * LSEQ + row)) * DMODEL + col] =
            val + cbias[col];
      }
    }
  }
}

// ---------------------------------------------------------------------------
// Flash attention: one wave handles 16 q rows of one (b,h).
// Online softmax over 32-key chunks; bias pre-softmax; mask post-softmax.
// P (probabilities) converted C-layout -> A-fragment layout via LDS.
// bias/mask are streamed once -> non-temporal loads so the reused bf16 k/v
// tensors (16 MB total) stay resident in the 192 MB L2.
// wavg written f32 as [B, Lq, H*64].
// ---------------------------------------------------------------------------
__global__ __launch_bounds__(128)
void attn_kernel(const unsigned* __restrict__ qp,   // [B,H,L,64] bf16 (u32 pairs)
                 const unsigned* __restrict__ kp,   // [B,H,L,64] bf16
                 const unsigned* __restrict__ vt,   // [B,H,64,L] bf16 (transposed)
                 const float* __restrict__ bias,    // [B,H,L,L] f32
                 const unsigned char* __restrict__ mask, // [B,H,L,L] bool
                 float* __restrict__ wavg)          // [B,L,1024] f32
{
  __shared__ unsigned Pbuf[4 * 16 * 16];            // per-wave 16x32 bf16 tile

  const int h = blockIdx.y, b = blockIdx.z;
  const int bh = b * HD + h;
  const int t = threadIdx.x, wave = t >> 5, lane = t & 31;
  const int lh = lane & 15, hi = lane >> 4;
  const int qbase = blockIdx.x * 64 + wave * 16;

  // Q A-fragments for K-dim 0..31 (a0) and 32..63 (a1), kept in registers
  AFrag a0, a1;
  {
    const unsigned* qrow = qp + ((size_t)bh * LSEQ + (qbase + lh)) * 32;
    const int bpA = hi * 8;
    #pragma unroll
    for (int j = 0; j < 4; ++j) {
      a0.u[j]     = qrow[bpA + j];
      a0.u[j + 4] = qrow[bpA + 4 + j];
      a1.u[j]     = qrow[16 + bpA + j];
      a1.u[j + 4] = qrow[16 + bpA + 4 + j];
    }
  }

  CAcc O[4];
  #pragma unroll
  for (int c = 0; c < 4; ++c)
    #pragma unroll
    for (int v = 0; v < 8; ++v) O[c].f[v] = 0.0f;
  float m[8], lsum[8];
  #pragma unroll
  for (int v = 0; v < 8; ++v) { m[v] = -1e30f; lsum[v] = 0.0f; }

  for (int ch = 0; ch < LSEQ / 32; ++ch) {
    // ---- S = q . k^T for 16x32 keys (two 16x16 C tiles) ----
    CAcc s[2];
    #pragma unroll
    for (int j = 0; j < 2; ++j) {
      #pragma unroll
      for (int v = 0; v < 8; ++v) s[j].f[v] = 0.0f;
      const int kvrow = ch * 32 + j * 16 + lh;
      const unsigned* krow = kp + ((size_t)bh * LSEQ + kvrow) * 32;
      AFrag bk;
      #pragma unroll
      for (int ti = 0; ti < 8; ++ti) bk.u[ti] = krow[hi * 8 + ti];
      s[j].v = __builtin_amdgcn_wmma_f32_16x16x32_bf16(
          false, a0.v, false, bk.v, (short)0, s[j].v, false, false);
      #pragma unroll
      for (int ti = 0; ti < 8; ++ti) bk.u[ti] = krow[16 + hi * 8 + ti];
      s[j].v = __builtin_amdgcn_wmma_f32_16x16x32_bf16(
          false, a1.v, false, bk.v, (short)0, s[j].v, false, false);
    }

    // ---- bias add + online softmax + post-softmax mask, stage P in LDS ----
    #pragma unroll
    for (int v = 0; v < 8; ++v) {
      const size_t brow =
          ((size_t)bh * LSEQ + (qbase + v + hi * 8)) * LSEQ + (size_t)ch * 32;
      s[0].f[v] += __builtin_nontemporal_load(bias + brow + lh);
      s[1].f[v] += __builtin_nontemporal_load(bias + brow + 16 + lh);
      float tmx = fmaxf(s[0].f[v], s[1].f[v]);
      #pragma unroll
      for (int off = 1; off < 16; off <<= 1)
        tmx = fmaxf(tmx, __shfl_xor(tmx, off, 32));
      const float mn = fmaxf(m[v], tmx);
      const float alpha = __expf(m[v] - mn);
      m[v] = mn;
      const float p0 = __expf(s[0].f[v] - mn);
      const float p1 = __expf(s[1].f[v] - mn);
      float rs = p0 + p1;
      #pragma unroll
      for (int off = 1; off < 16; off <<= 1) rs += __shfl_xor(rs, off, 32);
      lsum[v] = lsum[v] * alpha + rs;
      #pragma unroll
      for (int c = 0; c < 4; ++c) O[c].f[v] *= alpha;
      // masked_fill AFTER softmax numerator (denominator keeps all keys)
      const unsigned char mk0 = __builtin_nontemporal_load(mask + brow + lh);
      const unsigned char mk1 = __builtin_nontemporal_load(mask + brow + 16 + lh);
      const unsigned short h0 = mk0 ? f2bf(p0) : (unsigned short)0;
      const unsigned short h1 = mk1 ? f2bf(p1) : (unsigned short)0;
      ((unsigned short*)Pbuf)[(wave * 16 + v + hi * 8) * 32 + lh]      = h0;
      ((unsigned short*)Pbuf)[(wave * 16 + v + hi * 8) * 32 + 16 + lh] = h1;
    }
    __syncthreads();

    // ---- O += P . V  (P as A-fragment from LDS, V^T contiguous gather) ----
    AFrag pa;
    #pragma unroll
    for (int j = 0; j < 4; ++j) {
      pa.u[j]     = Pbuf[(wave * 16 + lh) * 16 + hi * 8 + j];
      pa.u[j + 4] = Pbuf[(wave * 16 + lh) * 16 + hi * 8 + 4 + j];
    }
    #pragma unroll
    for (int c = 0; c < 4; ++c) {
      const int dcol = c * 16 + lh;
      const unsigned* vcol =
          vt + (size_t)(bh * 64 + dcol) * (LSEQ / 2) + (size_t)ch * 16;
      AFrag bv;
      #pragma unroll
      for (int ti = 0; ti < 8; ++ti) bv.u[ti] = vcol[hi * 8 + ti];
      O[c].v = __builtin_amdgcn_wmma_f32_16x16x32_bf16(
          false, pa.v, false, bv.v, (short)0, O[c].v, false, false);
    }
    __syncthreads();
  }

  // ---- normalize and write weighted_avg as [B, Lq, H*64] f32 ----
  #pragma unroll
  for (int c = 0; c < 4; ++c) {
    #pragma unroll
    for (int v = 0; v < 8; ++v) {
      const int row = qbase + v + hi * 8;
      const int col = h * 64 + c * 16 + lh;
      wavg[((size_t)(b * LSEQ + row)) * DMODEL + col] = O[c].f[v] / lsum[v];
    }
  }
}

// ---------------------------------------------------------------------------
extern "C" void kernel_launch(void* const* d_in, const int* in_sizes, int n_in,
                              void* d_out, int out_size, void* d_ws, size_t ws_size,
                              hipStream_t stream) {
  (void)in_sizes; (void)n_in; (void)out_size; (void)ws_size;
  const float*         Q    = (const float*)d_in[0];
  const float*         K    = (const float*)d_in[1];
  const float*         V    = (const float*)d_in[2];
  const float*         bias = (const float*)d_in[3];
  const unsigned char* mask = (const unsigned char*)d_in[4];  // jnp bool_
  const float*         qw   = (const float*)d_in[5];
  const float*         kw   = (const float*)d_in[6];
  const float*         vw   = (const float*)d_in[7];
  const float*         gw   = (const float*)d_in[8];
  const float*         gb   = (const float*)d_in[9];
  const float*         ow   = (const float*)d_in[10];
  const float*         ob   = (const float*)d_in[11];

  // Workspace layout (bytes)
  char* ws = (char*)d_ws;
  unsigned short* qp = (unsigned short*)(ws + 0);          //  8 MB bf16 [B,H,L,64]
  unsigned short* kp = (unsigned short*)(ws + 8388608);    //  8 MB bf16 [B,H,L,64]
  unsigned short* vt = (unsigned short*)(ws + 16777216);   //  8 MB bf16 [B,H,64,L]
  float*          gate = (float*)(ws + 25165824);          // 16 MB f32 [B,L,1024]
  float*          wavg = (float*)(ws + 41943040);          // 16 MB f32 [B,L,1024]

  const dim3 bb(128);
  const dim3 gg(DMODEL / 64, LSEQ / 64, BATCH);
  // projections: q (scaled), k, v (transposed), gate (sigmoid + g_bias)
  gemm_proj<0><<<gg, bb, 0, stream>>>(Q, qw, nullptr, nullptr, qp, 0.125f);
  gemm_proj<0><<<gg, bb, 0, stream>>>(K, kw, nullptr, nullptr, kp, 1.0f);
  gemm_proj<2><<<gg, bb, 0, stream>>>(V, vw, nullptr, nullptr, vt, 1.0f);
  gemm_proj<1><<<gg, bb, 0, stream>>>(Q, gw, nullptr, gb,      gate, 1.0f);

  // attention (flash-style, online softmax, post-softmax mask)
  const dim3 ga(LSEQ / 64, HD, BATCH);
  attn_kernel<<<ga, bb, 0, stream>>>((const unsigned*)qp, (const unsigned*)kp,
                                     (const unsigned*)vt, bias, mask, wavg);

  // output projection with gating: out = (wavg*gate) @ o_w + o_bias
  gemm_proj<3><<<gg, bb, 0, stream>>>(wavg, ow, gate, ob, d_out, 1.0f);
}
